// GAT_78451872628891
// MI455X (gfx1250) — compile-verified, matrix operand order checked by reading, not run on hardware
//
#include <hip/hip_runtime.h>
#include <hip/hip_bf16.h>

#define N_NODES 4096
#define DIN     512
#define HDIM    128
#define KHEADS  4
#define DOUTC   16
#define ALPHA   0.2f

typedef __attribute__((ext_vector_type(16))) __bf16 v16bf;
typedef __attribute__((ext_vector_type(8)))  __bf16 v8bf;
typedef __attribute__((ext_vector_type(8)))  float  v8f;
typedef __attribute__((ext_vector_type(4)))  float  v4f;

__device__ __forceinline__ __bf16 f2bf(float f) {
  union { float f; unsigned u; } x; x.f = f;
  unsigned r = x.u + 0x7FFFu + ((x.u >> 16) & 1u);   // round-to-nearest-even
  unsigned short h = (unsigned short)(r >> 16);
  union { unsigned short s; __bf16 b; } y; y.s = h;
  return y.b;
}

// Load one 16x32 bf16 WMMA operand fragment from a row-major matrix (ld = row stride
// in elements). Works for A (rows = M) and for B when the matrix is stored transposed
// (rows = N columns of B). Per ISA 7.12.2: lanes 0-15 hold row=lane, K in {k0..k0+7,
// k0+16..k0+23}; lanes 16-31 hold row=lane-16, K shifted by +8.
__device__ __forceinline__ v16bf load_frag16x32(const __bf16* __restrict__ base, int ld,
                                                int row0, int k0, int lane) {
  int row = row0 + (lane & 15);
  int klo = k0 + ((lane & 16) ? 8 : 0);
  const __bf16* p = base + (size_t)row * ld + klo;
  v8bf lo = *(const v8bf*)(p);
  v8bf hi = *(const v8bf*)(p + 16);
  v16bf a;
#pragma unroll
  for (int e = 0; e < 8; ++e) { a[e] = lo[e]; a[8 + e] = hi[e]; }
  return a;
}

// Build the 16x32 probability A-fragment for one attention j-step.
// Vectorized: adj/dst come in as 4x float4 (b128) loads per lane.
__device__ __forceinline__ v16bf build_prob_frag(const float* __restrict__ arow,
                                                 const float* __restrict__ dstv_ptr,
                                                 int i, int j0, int jsh,
                                                 float si, float mi, float inv_s) {
  int base = j0 + jsh;                  // multiple of 8 -> 32B aligned
  const v4f* a4 = (const v4f*)(arow + base);
  const v4f* d4 = (const v4f*)(dstv_ptr + base);
  v4f av0 = a4[0], av1 = a4[1], av2 = a4[4], av3 = a4[5];   // j: +0..7, +16..23
  v4f dv0 = d4[0], dv1 = d4[1], dv2 = d4[4], dv3 = d4[5];
  float adjv[16], dstv[16];
#pragma unroll
  for (int q = 0; q < 4; ++q) {
    adjv[q] = av0[q]; adjv[4 + q] = av1[q]; adjv[8 + q] = av2[q]; adjv[12 + q] = av3[q];
    dstv[q] = dv0[q]; dstv[4 + q] = dv1[q]; dstv[8 + q] = dv2[q]; dstv[12 + q] = dv3[q];
  }
  v16bf a;
#pragma unroll
  for (int e = 0; e < 16; ++e) {
    int j = base + (e & 7) + ((e & 8) << 1);
    float p = 0.f;
    if (adjv[e] != 0.f || j == i) {
      float ev = si + dstv[e];
      ev = (ev >= 0.f) ? ev : ALPHA * ev;
      p = __expf(ev - mi) * inv_s;
    }
    a[e] = f2bf(p);
  }
  return a;
}

// ---------------- conversion kernels ----------------
__global__ void k_cvt_bf16(const float* __restrict__ src, __bf16* __restrict__ dst, int n) {
  int i = blockIdx.x * blockDim.x + threadIdx.x;
  if (i < n) dst[i] = f2bf(src[i]);
}

// W1 (K, DIN, H) -> W1bT (K, H, DIN)  (B operand, transposed)
__global__ void k_cvt_w1t(const float* __restrict__ W1, __bf16* __restrict__ W1bT) {
  int i = blockIdx.x * blockDim.x + threadIdx.x;       // K*H*DIN
  int d = i % DIN;
  int h = (i / DIN) % HDIM;
  int k = i / (DIN * HDIM);
  W1bT[i] = f2bf(W1[((size_t)k * DIN + d) * HDIM + h]);
}

// W2 (K*H, DOUT) -> W2bT (DOUT, K*H)
__global__ void k_cvt_w2t(const float* __restrict__ W2, __bf16* __restrict__ W2bT) {
  int i = blockIdx.x * blockDim.x + threadIdx.x;       // DOUT * (K*H)
  int d = i % (KHEADS * HDIM);
  int c = i / (KHEADS * HDIM);
  W2bT[i] = f2bf(W2[(size_t)d * DOUTC + c]);
}

// ---------------- GEMM1: Wh1[k] = X @ W1[k]  (bf16 WMMA, fp32 accum) ----------------
__global__ void k_gemm1(const __bf16* __restrict__ Xb, const __bf16* __restrict__ W1bT,
                        float* __restrict__ Wh1) {
  int wave = (int)((blockIdx.x * blockDim.x + threadIdx.x) >> 5);
  int lane = threadIdx.x & 31;
  int ct = wave % (HDIM / 16);
  int rt = (wave / (HDIM / 16)) % (N_NODES / 16);
  int k  = wave / ((HDIM / 16) * (N_NODES / 16));
  const __bf16* Bt = W1bT + (size_t)k * HDIM * DIN;
  int row0 = rt * 16, col0 = ct * 16;
  v8f acc = {};
  for (int k0 = 0; k0 < DIN; k0 += 32) {
    v16bf a = load_frag16x32(Xb, DIN, row0, k0, lane);
    v16bf b = load_frag16x32(Bt, DIN, col0, k0, lane);
    acc = __builtin_amdgcn_wmma_f32_16x16x32_bf16(false, a, false, b, (short)0, acc,
                                                  false, false);
  }
  float* C = Wh1 + (size_t)k * N_NODES * HDIM;
  int col   = col0 + (lane & 15);
  int rbase = row0 + ((lane & 16) ? 8 : 0);
#pragma unroll
  for (int r = 0; r < 8; ++r) C[(size_t)(rbase + r) * HDIM + col] = acc[r];
}

// ---------------- attention logits + Wh1^T (bf16) ----------------
__global__ void k_srcdst1(const float* __restrict__ Wh1, const float* __restrict__ a_src1,
                          const float* __restrict__ a_dst1, float* __restrict__ src1,
                          float* __restrict__ dst1, __bf16* __restrict__ Wh1bT) {
  int i = blockIdx.x * blockDim.x + threadIdx.x;       // K*N
  int n = i % N_NODES;
  int k = i / N_NODES;
  const float* row = Wh1 + ((size_t)k * N_NODES + n) * HDIM;
  const float* as  = a_src1 + k * HDIM;
  const float* ad  = a_dst1 + k * HDIM;
  __bf16* T = Wh1bT + (size_t)k * HDIM * N_NODES;
  float s = 0.f, d = 0.f;
  for (int h = 0; h < HDIM; ++h) {
    float v = row[h];
    s += v * as[h];
    d += v * ad[h];
    T[(size_t)h * N_NODES + n] = f2bf(v);
  }
  src1[i] = s; dst1[i] = d;
}

// ---------------- masked row max / sum(exp) pre-pass ----------------
__global__ void k_rowstats(const float* __restrict__ adj, const float* __restrict__ src,
                           const float* __restrict__ dst, float* __restrict__ m_out,
                           float* __restrict__ s_out) {
  int bid = blockIdx.x;            // kh*N + i  (kh=0 for layer 2 when grid == N)
  int i   = bid % N_NODES;
  int kh  = bid / N_NODES;
  float si = src[bid];
  const float* dstk = dst + (size_t)kh * N_NODES;
  const float* arow = adj + (size_t)i * N_NODES;
  __shared__ float red[256];
  float m = -3.0e38f;
  for (int j = threadIdx.x; j < N_NODES; j += 256) {
    if (arow[j] != 0.f || j == i) {
      float e = si + dstk[j];
      e = (e >= 0.f) ? e : ALPHA * e;
      m = fmaxf(m, e);
    }
  }
  red[threadIdx.x] = m; __syncthreads();
  for (int o = 128; o > 0; o >>= 1) {
    if (threadIdx.x < o) red[threadIdx.x] = fmaxf(red[threadIdx.x], red[threadIdx.x + o]);
    __syncthreads();
  }
  m = red[0]; __syncthreads();
  float s = 0.f;
  for (int j = threadIdx.x; j < N_NODES; j += 256) {
    if (arow[j] != 0.f || j == i) {
      float e = si + dstk[j];
      e = (e >= 0.f) ? e : ALPHA * e;
      s += __expf(e - m);
    }
  }
  red[threadIdx.x] = s; __syncthreads();
  for (int o = 128; o > 0; o >>= 1) {
    if (threadIdx.x < o) red[threadIdx.x] += red[threadIdx.x + o];
    __syncthreads();
  }
  if (threadIdx.x == 0) { m_out[bid] = m; s_out[bid] = red[0]; }
}

// ---------------- layer-1 aggregation: out1[k] = softmax(e) @ Wh1[k] ----------------
// One wave per (head, 16-row tile); 8 accumulators cover all H=128 output columns.
__global__ void k_agg1(const float* __restrict__ adj, const float* __restrict__ src1,
                       const float* __restrict__ dst1, const float* __restrict__ m1,
                       const float* __restrict__ s1, const __bf16* __restrict__ Wh1bT,
                       float* __restrict__ out1) {
  int wave = (int)((blockIdx.x * blockDim.x + threadIdx.x) >> 5);
  int lane = threadIdx.x & 31;
  int rt = wave % (N_NODES / 16);
  int k  = wave / (N_NODES / 16);
  int i  = rt * 16 + (lane & 15);
  float si    = src1[k * N_NODES + i];
  float mi    = m1[k * N_NODES + i];
  float inv_s = 1.0f / s1[k * N_NODES + i];
  const float* dstk  = dst1 + (size_t)k * N_NODES;
  const float* arow  = adj + (size_t)i * N_NODES;
  const __bf16* Bt   = Wh1bT + (size_t)k * HDIM * N_NODES;
  v8f acc[8];
#pragma unroll
  for (int c = 0; c < 8; ++c) acc[c] = (v8f){};
  int jsh = (lane & 16) ? 8 : 0;
  for (int j0 = 0; j0 < N_NODES; j0 += 32) {
    if (j0 + 32 < N_NODES)
      __builtin_prefetch(arow + j0 + 32 + jsh, 0, 0);   // global_prefetch of next chunk
    v16bf a = build_prob_frag(arow, dstk, i, j0, jsh, si, mi, inv_s);
#pragma unroll
    for (int c = 0; c < 8; ++c) {
      v16bf b = load_frag16x32(Bt, N_NODES, c * 16, j0, lane);
      acc[c] = __builtin_amdgcn_wmma_f32_16x16x32_bf16(false, a, false, b, (short)0,
                                                       acc[c], false, false);
    }
  }
  float* C = out1 + (size_t)k * N_NODES * HDIM;
  int rbase = rt * 16 + ((lane & 16) ? 8 : 0);
  int coll  = lane & 15;
#pragma unroll
  for (int c = 0; c < 8; ++c)
#pragma unroll
    for (int r = 0; r < 8; ++r)
      C[(size_t)(rbase + r) * HDIM + c * 16 + coll] = acc[c][r];
}

// ---------------- ELU + concat heads -> bf16 h1 ----------------
__global__ void k_elu_cvt(const float* __restrict__ out1, __bf16* __restrict__ h1b) {
  int idx = blockIdx.x * blockDim.x + threadIdx.x;     // [k][n][h]
  int h = idx % HDIM;
  int n = (idx / HDIM) % N_NODES;
  int k = idx / (HDIM * N_NODES);
  float v = out1[idx];
  v = (v >= 0.f) ? v : (__expf(v) - 1.f);
  h1b[(size_t)n * (KHEADS * HDIM) + k * HDIM + h] = f2bf(v);
}

// ---------------- GEMM2: Wh2 = h1 @ W2 ----------------
__global__ void k_gemm2(const __bf16* __restrict__ h1b, const __bf16* __restrict__ W2bT,
                        float* __restrict__ Wh2) {
  int wave = (int)((blockIdx.x * blockDim.x + threadIdx.x) >> 5);
  int lane = threadIdx.x & 31;
  int row0 = wave * 16;
  v8f acc = {};
  for (int k0 = 0; k0 < KHEADS * HDIM; k0 += 32) {
    v16bf a = load_frag16x32(h1b, KHEADS * HDIM, row0, k0, lane);
    v16bf b = load_frag16x32(W2bT, KHEADS * HDIM, 0, k0, lane);
    acc = __builtin_amdgcn_wmma_f32_16x16x32_bf16(false, a, false, b, (short)0, acc,
                                                  false, false);
  }
  int col   = lane & 15;
  int rbase = row0 + ((lane & 16) ? 8 : 0);
#pragma unroll
  for (int r = 0; r < 8; ++r) Wh2[(size_t)(rbase + r) * DOUTC + col] = acc[r];
}

// ---------------- layer-2 logits + Wh2^T bf16 ----------------
__global__ void k_srcdst2(const float* __restrict__ Wh2, const float* __restrict__ a_src2,
                          const float* __restrict__ a_dst2, float* __restrict__ src2,
                          float* __restrict__ dst2, __bf16* __restrict__ Wh2bT) {
  int n = blockIdx.x * blockDim.x + threadIdx.x;
  if (n >= N_NODES) return;
  float s = 0.f, d = 0.f;
  for (int c = 0; c < DOUTC; ++c) {
    float v = Wh2[(size_t)n * DOUTC + c];
    s += v * a_src2[c];
    d += v * a_dst2[c];
    Wh2bT[(size_t)c * N_NODES + n] = f2bf(v);
  }
  src2[n] = s; dst2[n] = d;
}

// ---------------- layer-2 aggregation (single head, 16 cols) ----------------
__global__ void k_agg2(const float* __restrict__ adj, const float* __restrict__ src2,
                       const float* __restrict__ dst2, const float* __restrict__ m2,
                       const float* __restrict__ s2, const __bf16* __restrict__ Wh2bT,
                       float* __restrict__ h2) {
  int wave = (int)((blockIdx.x * blockDim.x + threadIdx.x) >> 5);
  int lane = threadIdx.x & 31;
  int i = wave * 16 + (lane & 15);
  float si    = src2[i];
  float mi    = m2[i];
  float inv_s = 1.0f / s2[i];
  const float* arow = adj + (size_t)i * N_NODES;
  v8f acc = {};
  int jsh = (lane & 16) ? 8 : 0;
  for (int j0 = 0; j0 < N_NODES; j0 += 32) {
    if (j0 + 32 < N_NODES)
      __builtin_prefetch(arow + j0 + 32 + jsh, 0, 0);
    v16bf a = build_prob_frag(arow, dst2, i, j0, jsh, si, mi, inv_s);
    v16bf b = load_frag16x32(Wh2bT, N_NODES, 0, j0, lane);
    acc = __builtin_amdgcn_wmma_f32_16x16x32_bf16(false, a, false, b, (short)0, acc,
                                                  false, false);
  }
  int rbase = wave * 16 + ((lane & 16) ? 8 : 0);
  int col   = lane & 15;
#pragma unroll
  for (int r = 0; r < 8; ++r) h2[(size_t)(rbase + r) * DOUTC + col] = acc[r];
}

// ---------------- log_softmax over 16-wide rows ----------------
__global__ void k_logsoftmax(const float* __restrict__ h2, float* __restrict__ out) {
  int n = blockIdx.x * blockDim.x + threadIdx.x;
  if (n >= N_NODES) return;
  const float* r = h2 + (size_t)n * DOUTC;
  float m = r[0];
#pragma unroll
  for (int c = 1; c < DOUTC; ++c) m = fmaxf(m, r[c]);
  float s = 0.f;
#pragma unroll
  for (int c = 0; c < DOUTC; ++c) s += __expf(r[c] - m);
  float lse = m + __logf(s);
#pragma unroll
  for (int c = 0; c < DOUTC; ++c) out[(size_t)n * DOUTC + c] = r[c] - lse;
}

extern "C" void kernel_launch(void* const* d_in, const int* in_sizes, int n_in,
                              void* d_out, int out_size, void* d_ws, size_t ws_size,
                              hipStream_t stream) {
  const float* X      = (const float*)d_in[0];
  const float* adj    = (const float*)d_in[1];
  const float* W1     = (const float*)d_in[2];
  const float* a_src1 = (const float*)d_in[3];
  const float* a_dst1 = (const float*)d_in[4];
  const float* W2     = (const float*)d_in[5];
  const float* a_src2 = (const float*)d_in[6];
  const float* a_dst2 = (const float*)d_in[7];
  float* out = (float*)d_out;

  char* w = (char*)d_ws;
  auto alloc = [&](size_t bytes) -> void* {
    void* p = (void*)w;
    w += (bytes + 255) & ~(size_t)255;
    return p;
  };
  __bf16* Xb     = (__bf16*)alloc((size_t)N_NODES * DIN * 2);
  __bf16* W1bT   = (__bf16*)alloc((size_t)KHEADS * HDIM * DIN * 2);
  __bf16* W2bT   = (__bf16*)alloc((size_t)DOUTC * KHEADS * HDIM * 2);
  float*  Wh1    = (float*) alloc((size_t)KHEADS * N_NODES * HDIM * 4);
  __bf16* Wh1bT  = (__bf16*)alloc((size_t)KHEADS * HDIM * N_NODES * 2);
  float*  src1   = (float*) alloc((size_t)KHEADS * N_NODES * 4);
  float*  dst1   = (float*) alloc((size_t)KHEADS * N_NODES * 4);
  float*  m1     = (float*) alloc((size_t)KHEADS * N_NODES * 4);
  float*  s1     = (float*) alloc((size_t)KHEADS * N_NODES * 4);
  float*  out1   = (float*) alloc((size_t)KHEADS * N_NODES * HDIM * 4);
  __bf16* h1b    = (__bf16*)alloc((size_t)N_NODES * KHEADS * HDIM * 2);
  float*  Wh2    = (float*) alloc((size_t)N_NODES * DOUTC * 4);
  __bf16* Wh2bT  = (__bf16*)alloc((size_t)DOUTC * N_NODES * 2);
  float*  src2   = (float*) alloc((size_t)N_NODES * 4);
  float*  dst2   = (float*) alloc((size_t)N_NODES * 4);
  float*  m2     = (float*) alloc((size_t)N_NODES * 4);
  float*  s2     = (float*) alloc((size_t)N_NODES * 4);
  float*  h2     = (float*) alloc((size_t)N_NODES * DOUTC * 4);

  // conversions
  k_cvt_bf16<<<(N_NODES * DIN) / 256, 256, 0, stream>>>(X, Xb, N_NODES * DIN);
  k_cvt_w1t<<<(KHEADS * HDIM * DIN) / 256, 256, 0, stream>>>(W1, W1bT);
  k_cvt_w2t<<<(DOUTC * KHEADS * HDIM) / 256, 256, 0, stream>>>(W2, W2bT);

  // layer 1
  {
    int waves = KHEADS * (N_NODES / 16) * (HDIM / 16);   // 8192
    k_gemm1<<<waves / 4, 128, 0, stream>>>(Xb, W1bT, Wh1);
  }
  k_srcdst1<<<(KHEADS * N_NODES) / 256, 256, 0, stream>>>(Wh1, a_src1, a_dst1, src1, dst1, Wh1bT);
  k_rowstats<<<KHEADS * N_NODES, 256, 0, stream>>>(adj, src1, dst1, m1, s1);
  {
    int waves = KHEADS * (N_NODES / 16);                 // 1024
    k_agg1<<<waves / 4, 128, 0, stream>>>(adj, src1, dst1, m1, s1, Wh1bT, out1);
  }
  k_elu_cvt<<<(KHEADS * N_NODES * HDIM) / 256, 256, 0, stream>>>(out1, h1b);

  // layer 2
  {
    int waves = N_NODES / 16;                            // 256
    k_gemm2<<<waves / 4, 128, 0, stream>>>(h1b, W2bT, Wh2);
  }
  k_srcdst2<<<N_NODES / 256, 256, 0, stream>>>(Wh2, a_src2, a_dst2, src2, dst2, Wh2bT);
  k_rowstats<<<N_NODES, 256, 0, stream>>>(adj, src2, dst2, m2, s2);
  {
    int waves = N_NODES / 16;                            // 256
    k_agg2<<<waves / 4, 128, 0, stream>>>(adj, src2, dst2, m2, s2, Wh2bT, h2);
  }
  k_logsoftmax<<<N_NODES / 256, 256, 0, stream>>>(h2, out);
  (void)in_sizes; (void)n_in; (void)out_size; (void)ws_size;
}
